// SVRSheafNet_60997125537950
// MI455X (gfx1250) — matrix operand order repaired and checked by reading, not compiled, not run on hardware
//
#include <hip/hip_runtime.h>
#include <hip/hip_bf16.h>

// ---------------------------------------------------------------------------
// SVRSheafNet forward on gfx1250 (MI455X).
// GEMMs via v_wmma_f32_16x16x32_bf16 with split-bf16 (hi+lo) error compensation.
// Sheaf-Laplacian SpMVs via L2-resident gathers + global_atomic_add_f32 scatters.
// Requires ws_size >= ~220 MB.
// ---------------------------------------------------------------------------

#define NNODES 50000
#define NEDGES 512000
#define HDIM   128
#define DT_    0.1f
#define EPS_   1e-3f

typedef __attribute__((ext_vector_type(16))) __bf16 bf16x16;
typedef __attribute__((ext_vector_type(8)))  float  f32x8;

union Frag { bf16x16 v; unsigned int u[8]; };

__device__ __forceinline__ unsigned short f2bf(float f) {
    unsigned u = __float_as_uint(f);
    u += 0x7FFFu + ((u >> 16) & 1u);
    return (unsigned short)(u >> 16);
}
__device__ __forceinline__ float bf2f(unsigned short h) {
    return __uint_as_float(((unsigned)h) << 16);
}
// order-preserving float->uint key for atomicMax on floats
__device__ __forceinline__ unsigned fkey(float f) {
    unsigned u = __float_as_uint(f);
    return (u & 0x80000000u) ? ~u : (u | 0x80000000u);
}
__device__ __forceinline__ float funkey(unsigned k) {
    unsigned u = (k & 0x80000000u) ? (k & 0x7fffffffu) : ~k;
    return __uint_as_float(u);
}
__device__ __forceinline__ float sigm(float x) { return 1.0f / (1.0f + expf(-x)); }
__device__ __forceinline__ float lrelu(float x) { return x > 0.0f ? x : 0.2f * x; }

// ---------------------------------------------------------------------------
// Weight prep: f32 [K,NOUT] -> swizzled split-bf16 B-fragments.
// Layout: Wsw[((kt*NT+nt)*32+lane)*16+e], hi at 0, lo at +K*NOUT.
// element e of lane: Kg = kt*32 + (lane>>4)*8 + (e&7) + (e>>3)*16 ; col = nt*16 + (lane&15)
// ---------------------------------------------------------------------------
__global__ void k_prep_w(const float* __restrict__ W, unsigned short* __restrict__ Wsw,
                         int K, int NOUT) {
    int ntiles = NOUT >> 4;
    int total = K * NOUT;
    int gid = blockIdx.x * blockDim.x + threadIdx.x;
    if (gid >= total) return;
    int e = gid & 15;
    int lane = (gid >> 4) & 31;
    int tile = gid >> 9;
    int nt = tile % ntiles;
    int kt = tile / ntiles;
    int Kg = kt * 32 + (lane >> 4) * 8 + (e & 7) + (e >> 3) * 16;
    int col = nt * 16 + (lane & 15);
    float v = W[(size_t)Kg * NOUT + col];
    unsigned short h = f2bf(v);
    Wsw[gid] = h;
    Wsw[total + gid] = f2bf(v - bf2f(h));
}

// ---------------------------------------------------------------------------
// WMMA GEMM: out[M,NOUT] = A[M,K] @ W (+bias). NOUT = NTILES*16, K%32==0.
// 4 waves / block, one 16-row M-tile per wave. Split-bf16: 3 WMMAs per k-tile.
// LNSIG epilogue: +bias, layernorm(g,b), sigmoid (for the input projection).
// ---------------------------------------------------------------------------
template <int NTILES, bool LNSIG>
__global__ void k_gemm_wmma(const float* __restrict__ A,
                            const unsigned short* __restrict__ Wsw,
                            const float* __restrict__ bias,
                            const float* __restrict__ ln_g,
                            const float* __restrict__ ln_b,
                            float* __restrict__ out, int mtiles, int K) {
    __shared__ float smem[LNSIG ? (4 * 16 * 128) : 4];
    const int wv = threadIdx.x >> 5;
    const int lane = threadIdx.x & 31;
    const int half = lane >> 4;
    const int l16 = lane & 15;
    const int mtile = blockIdx.x * 4 + wv;
    const int ktiles = K >> 5;
    const int total = K * (NTILES * 16);

    if (mtile < mtiles) {
        f32x8 acc[NTILES];
#pragma unroll
        for (int n = 0; n < NTILES; ++n)
#pragma unroll
            for (int q = 0; q < 8; ++q) acc[n][q] = 0.0f;

        const float* arow = A + (size_t)(mtile * 16 + l16) * K;
        for (int kt = 0; kt < ktiles; ++kt) {
            // ---- A fragment (16-bit A layout: e<8 -> K=half*8+e ; e>=8 -> K=16+half*8+e-8)
            const float* p0 = arow + kt * 32 + half * 8;
            float4 a0 = *(const float4*)(p0);
            float4 a1 = *(const float4*)(p0 + 4);
            float4 a2 = *(const float4*)(p0 + 16);
            float4 a3 = *(const float4*)(p0 + 20);
            float av[16] = {a0.x, a0.y, a0.z, a0.w, a1.x, a1.y, a1.z, a1.w,
                            a2.x, a2.y, a2.z, a2.w, a3.x, a3.y, a3.z, a3.w};
            Frag ahi, alo;
#pragma unroll
            for (int p = 0; p < 8; ++p) {
                float v0 = av[2 * p], v1 = av[2 * p + 1];
                unsigned short h0 = f2bf(v0), h1 = f2bf(v1);
                ahi.u[p] = (unsigned)h0 | ((unsigned)h1 << 16);
                unsigned short e0 = f2bf(v0 - bf2f(h0)), e1 = f2bf(v1 - bf2f(h1));
                alo.u[p] = (unsigned)e0 | ((unsigned)e1 << 16);
            }
#pragma unroll
            for (int nt = 0; nt < NTILES; ++nt) {
                const unsigned short* wb =
                    Wsw + ((size_t)(kt * NTILES + nt) * 32 + lane) * 16;
                Frag bhi, blo;
                const uint4* qh = (const uint4*)wb;
                uint4 h0v = qh[0], h1v = qh[1];
                bhi.u[0] = h0v.x; bhi.u[1] = h0v.y; bhi.u[2] = h0v.z; bhi.u[3] = h0v.w;
                bhi.u[4] = h1v.x; bhi.u[5] = h1v.y; bhi.u[6] = h1v.z; bhi.u[7] = h1v.w;
                const uint4* ql = (const uint4*)(wb + total);
                uint4 l0v = ql[0], l1v = ql[1];
                blo.u[0] = l0v.x; blo.u[1] = l0v.y; blo.u[2] = l0v.z; blo.u[3] = l0v.w;
                blo.u[4] = l1v.x; blo.u[5] = l1v.y; blo.u[6] = l1v.z; blo.u[7] = l1v.w;
                acc[nt] = __builtin_amdgcn_wmma_f32_16x16x32_bf16(
                    false, ahi.v, false, bhi.v, (short)0, acc[nt], false, false);
                acc[nt] = __builtin_amdgcn_wmma_f32_16x16x32_bf16(
                    false, ahi.v, false, blo.v, (short)0, acc[nt], false, false);
                acc[nt] = __builtin_amdgcn_wmma_f32_16x16x32_bf16(
                    false, alo.v, false, bhi.v, (short)0, acc[nt], false, false);
            }
        }
        // ---- epilogue (C layout: row = half*8+r, col = nt*16+l16)
        if constexpr (LNSIG) {
#pragma unroll
            for (int nt = 0; nt < NTILES; ++nt)
#pragma unroll
                for (int r = 0; r < 8; ++r)
                    smem[(wv * 16 + half * 8 + r) * 128 + nt * 16 + l16] =
                        acc[nt][r] + bias[nt * 16 + l16];
        } else {
#pragma unroll
            for (int nt = 0; nt < NTILES; ++nt)
#pragma unroll
                for (int r = 0; r < 8; ++r) {
                    float v = acc[nt][r];
                    if (bias) v += bias[nt * 16 + l16];
                    out[(size_t)(mtile * 16 + half * 8 + r) * (NTILES * 16) +
                        nt * 16 + l16] = v;
                }
        }
    }
    if constexpr (LNSIG) {
        __syncthreads();
        int t = threadIdx.x;
        if (t < 64) {
            int w2 = t >> 4, rl = t & 15;
            int gm = blockIdx.x * 4 + w2;
            if (gm < mtiles) {
                const float* rowp = &smem[(w2 * 16 + rl) * 128];
                float s = 0.f, s2 = 0.f;
                for (int j = 0; j < 128; ++j) {
                    float v = rowp[j]; s += v; s2 += v * v;
                }
                float mean = s * (1.0f / 128.0f);
                float var = s2 * (1.0f / 128.0f) - mean * mean;
                float rstd = rsqrtf(var + 1e-5f);
                float* op = out + (size_t)(gm * 16 + rl) * 128;
                for (int j = 0; j < 128; ++j) {
                    float y = (rowp[j] - mean) * rstd * ln_g[j] + ln_b[j];
                    op[j] = sigm(y);
                }
            }
        }
    }
}

// ---------------------------------------------------------------------------
// Elementwise / reduction / edge kernels
// ---------------------------------------------------------------------------
__global__ void k_zero(float* p, int n) {
    int i = blockIdx.x * blockDim.x + threadIdx.x;
    if (i < n) p[i] = 0.0f;
}
__global__ void k_flags_init(int* f) {
    f[0] = 0; f[1] = 0; f[2] = 0;
}
// wL + deg: one wave32 per edge; wL = w^2 * mean(feat_row^2)
__global__ void k_wl_deg(const float* __restrict__ hW, const int* __restrict__ row,
                         const int* __restrict__ col, float* __restrict__ wL,
                         float* __restrict__ deg, int nE) {
    int t = blockIdx.x * blockDim.x + threadIdx.x;
    int e = t >> 5, lane = t & 31;
    if (e >= nE) return;
    int r = row[e], c = col[e];
    float4 a = ((const float4*)(hW + (size_t)r * HDIM))[lane];
    float4 b = ((const float4*)(hW + (size_t)c * HDIM))[lane];
    float d0 = a.x - b.x, d1 = a.y - b.y, d2 = a.z - b.z, d3 = a.w - b.w;
    float sd = d0 * d0 + d1 * d1 + d2 * d2 + d3 * d3;
    float sf = a.x * a.x + a.y * a.y + a.z * a.z + a.w * a.w;
    for (int off = 16; off; off >>= 1) {
        sd += __shfl_down(sd, off);
        sf += __shfl_down(sf, off);
    }
    if (lane == 0) {
        float C = sd * (1.0f / 128.0f);
        float ex = expf(-C * (1.0f / EPS_));
        float P0 = fminf(fmaxf(ex, 1e-3f), 1.0f);
        float Ps = fminf(fmaxf((P0 + 1e-12f) * ex, 1e-3f), 1.0f);
        float w = 0.7f * P0 + 0.3f * Ps;
        float wl = w * w * sf * (1.0f / 128.0f);
        wL[e] = wl;
        unsafeAtomicAdd(deg + r, wl);
        unsafeAtomicAdd(deg + c, wl);
    }
}
// D[i,j] = (a + b*deg[i]) * S[i,j]
__global__ void k_scale_deg(float* __restrict__ D, const float* __restrict__ S,
                            const float* __restrict__ deg, float a, float b,
                            const int* gate, int n) {
    if (*gate) return;
    int i = blockIdx.x * blockDim.x + threadIdx.x;
    if (i < n) D[i] = (a + b * deg[i >> 7]) * S[i];
}
// D[row] += sc*wL*S[col] ; D[col] += sc*wL*S[row]  (4 threads/edge, 32 cols each)
__global__ void k_edge_scatter(float* __restrict__ D, const float* __restrict__ S,
                               const float* __restrict__ wL, const int* __restrict__ row,
                               const int* __restrict__ col, float sc, const int* gate,
                               int nE) {
    if (*gate) return;
    int t = blockIdx.x * blockDim.x + threadIdx.x;
    if (t >= nE * 4) return;
    int e = t >> 2, part = t & 3;
    float w = sc * wL[e];
    int r = row[e], c = col[e];
    const float4* Sc = (const float4*)(S + (size_t)c * HDIM + part * 32);
    const float4* Sr = (const float4*)(S + (size_t)r * HDIM + part * 32);
    float* Dr = D + (size_t)r * HDIM + part * 32;
    float* Dc = D + (size_t)c * HDIM + part * 32;
#pragma unroll
    for (int q = 0; q < 8; ++q) {
        float4 vc = Sc[q], vr = Sr[q];
        unsafeAtomicAdd(Dr + q * 4 + 0, w * vc.x);
        unsafeAtomicAdd(Dr + q * 4 + 1, w * vc.y);
        unsafeAtomicAdd(Dr + q * 4 + 2, w * vc.z);
        unsafeAtomicAdd(Dr + q * 4 + 3, w * vc.w);
        unsafeAtomicAdd(Dc + q * 4 + 0, w * vr.x);
        unsafeAtomicAdd(Dc + q * 4 + 1, w * vr.y);
        unsafeAtomicAdd(Dc + q * 4 + 2, w * vr.z);
        unsafeAtomicAdd(Dc + q * 4 + 3, w * vr.w);
    }
}
// out[j] += sum_i A[i,j]*B[i,j]  (column-wise dot, out pre-zeroed)
__global__ void k_coldot(const float* __restrict__ A, const float* __restrict__ B,
                         float* __restrict__ out, int nrows, const int* gate) {
    if (*gate) return;
    __shared__ float part[256];
    int j = threadIdx.x & 127, sub = threadIdx.x >> 7;
    float s = 0.f;
    for (int r = blockIdx.x * 2 + sub; r < nrows; r += gridDim.x * 2)
        s += A[(size_t)r * 128 + j] * B[(size_t)r * 128 + j];
    part[threadIdx.x] = s;
    __syncthreads();
    if (sub == 0) unsafeAtomicAdd(&out[j], part[j] + part[128 + j]);
}
__global__ void k_cg_alpha(float* a, const float* rs, const float* den, const int* gate) {
    if (*gate) return;
    int t = threadIdx.x;
    a[t] = rs[t] / (den[t] + 1e-16f);
}
__global__ void k_cg_xr(float* __restrict__ X, float* __restrict__ R,
                        const float* __restrict__ P, const float* __restrict__ AP,
                        const float* __restrict__ a, const int* gate, int n) {
    if (*gate) return;
    int i = blockIdx.x * blockDim.x + threadIdx.x;
    if (i >= n) return;
    float al = a[i & 127];
    X[i] += al * P[i];
    R[i] -= al * AP[i];
}
__global__ void k_cg_finish(float* rs, const float* rsn, float* beta, int* f1, int* f2) {
    if (*f1) return;
    __shared__ float red[128];
    int t = threadIdx.x;
    red[t] = rsn[t];
    __syncthreads();
    for (int off = 64; off; off >>= 1) {
        if (t < off) red[t] = fmaxf(red[t], red[t + off]);
        __syncthreads();
    }
    float b = rsn[t] / (rs[t] + 1e-16f);
    beta[t] = b;
    rs[t] = rsn[t];
    if (t == 0 && red[0] < 1e-8f) { *f2 = 1; *f1 = 1; }
}
__global__ void k_cg_p(float* __restrict__ P, const float* __restrict__ R,
                       const float* __restrict__ beta, const int* gate, int n) {
    if (*gate) return;
    int i = blockIdx.x * blockDim.x + threadIdx.x;
    if (i < n) P[i] = R[i] + beta[i & 127] * P[i];
}
__global__ void k_isd(const float* deg, float* isd, int n) {
    int i = blockIdx.x * blockDim.x + threadIdx.x;
    if (i < n) isd[i] = sqrtf(1.0f / fmaxf(deg[i], 1e-8f));
}
__global__ void k_softmax4(const float* g, float* aq) {
    float m = fmaxf(fmaxf(g[0], g[1]), fmaxf(g[2], g[3]));
    float e0 = expf(g[0] - m), e1 = expf(g[1] - m), e2 = expf(g[2] - m), e3 = expf(g[3] - m);
    float s = e0 + e1 + e2 + e3;
    aq[0] = e0 / s; aq[1] = e1 / s; aq[2] = e2 / s; aq[3] = e3 / s;
}
__global__ void k_mul_isd(float* __restrict__ U, const float* __restrict__ M,
                          const float* __restrict__ isd, int n) {
    int i = blockIdx.x * blockDim.x + threadIdx.x;
    if (i < n) U[i] = isd[i >> 7] * M[i];
}
__global__ void k_tilde_out(float* __restrict__ T, const float* __restrict__ M,
                            const float* __restrict__ L, const float* __restrict__ isd, int n) {
    int i = blockIdx.x * blockDim.x + threadIdx.x;
    if (i < n) T[i] = M[i] + isd[i >> 7] * L[i];
}
__global__ void k_cheb_next(float* __restrict__ Tn, const float* __restrict__ Tc,
                            const float* __restrict__ L, const float* __restrict__ Tp,
                            const float* __restrict__ isd, int n) {
    int i = blockIdx.x * blockDim.x + threadIdx.x;
    if (i < n) Tn[i] = 2.0f * (Tc[i] + isd[i >> 7] * L[i]) - Tp[i];
}
__global__ void k_hacc_init(float* __restrict__ Hc, const float* __restrict__ h,
                            const float* __restrict__ T1, const float* aq, int n) {
    int i = blockIdx.x * blockDim.x + threadIdx.x;
    if (i < n) Hc[i] = aq[0] * h[i] + aq[1] * T1[i];
}
__global__ void k_hacc_add(float* __restrict__ Hc, const float* __restrict__ T,
                           const float* aq, int q, int n) {
    int i = blockIdx.x * blockDim.x + threadIdx.x;
    if (i < n) Hc[i] += aq[q] * T[i];
}
__global__ void k_fused(float* __restrict__ F, const float* __restrict__ h,
                        const float* __restrict__ X, const float* __restrict__ Hc,
                        const float* asvr, const float* aafm, int n) {
    int i = blockIdx.x * blockDim.x + threadIdx.x;
    if (i >= n) return;
    F[i] = h[i] + sigm(asvr[0]) * X[i] + sigm(aafm[0]) * Hc[i];
}
// per-(node,head) attention logits
__global__ void k_head_dots(const float* __restrict__ xw, const float* __restrict__ asrc,
                            const float* __restrict__ adst, float* __restrict__ es,
                            float* __restrict__ ed, int nN, int heads, int oc) {
    int t = blockIdx.x * blockDim.x + threadIdx.x;
    if (t >= nN * heads) return;
    int n = t / heads, hd = t - n * heads;
    const float* xp = xw + (size_t)n * heads * oc + hd * oc;
    float s = 0.f, d = 0.f;
    for (int c = 0; c < oc; ++c) {
        s += xp[c] * asrc[hd * oc + c];
        d += xp[c] * adst[hd * oc + c];
    }
    es[t] = s;
    ed[t] = d;
}
__global__ void k_init_att(unsigned* m, float* den, int n) {
    int i = blockIdx.x * blockDim.x + threadIdx.x;
    if (i < n) { m[i] = 0x007FFFFFu; den[i] = 0.0f; }  // fkey(-inf)
}
__global__ void k_att_max(const float* __restrict__ es, const float* __restrict__ ed,
                          const int* __restrict__ row, const int* __restrict__ col,
                          unsigned* __restrict__ m, int nE, int nN, int heads) {
    int t = blockIdx.x * blockDim.x + threadIdx.x;
    if (t >= (nE + nN) * heads) return;
    int idx = t / heads, hd = t - idx * heads;
    int s, d;
    if (idx < nE) { s = row[idx]; d = col[idx]; } else { s = d = idx - nE; }
    float e = lrelu(es[s * heads + hd] + ed[d * heads + hd]);
    atomicMax(&m[d * heads + hd], fkey(e));
}
__global__ void k_att_den(const float* __restrict__ es, const float* __restrict__ ed,
                          const int* __restrict__ row, const int* __restrict__ col,
                          const unsigned* __restrict__ m, float* __restrict__ den,
                          int nE, int nN, int heads) {
    int t = blockIdx.x * blockDim.x + threadIdx.x;
    if (t >= (nE + nN) * heads) return;
    int idx = t / heads, hd = t - idx * heads;
    int s, d;
    if (idx < nE) { s = row[idx]; d = col[idx]; } else { s = d = idx - nE; }
    float e = lrelu(es[s * heads + hd] + ed[d * heads + hd]);
    float ex = expf(e - funkey(m[d * heads + hd]));
    unsafeAtomicAdd(&den[d * heads + hd], ex);
}
__global__ void k_att_agg(const float* __restrict__ es, const float* __restrict__ ed,
                          const int* __restrict__ row, const int* __restrict__ col,
                          const unsigned* __restrict__ m, const float* __restrict__ den,
                          const float* __restrict__ xw, float* __restrict__ outacc,
                          int nE, int nN, int heads, int oc) {
    int t = blockIdx.x * blockDim.x + threadIdx.x;
    if (t >= (nE + nN) * heads) return;
    int idx = t / heads, hd = t - idx * heads;
    int s, d;
    if (idx < nE) { s = row[idx]; d = col[idx]; } else { s = d = idx - nE; }
    float e = lrelu(es[s * heads + hd] + ed[d * heads + hd]);
    float ex = expf(e - funkey(m[d * heads + hd]));
    float al = ex / (den[d * heads + hd] + 1e-16f);
    int hs = heads * oc;
    const float* xp = xw + (size_t)s * hs + hd * oc;
    float* op = outacc + (size_t)d * hs + hd * oc;
    for (int c = 0; c < oc; ++c) unsafeAtomicAdd(op + c, al * xp[c]);
}
__global__ void k_elu_bias(float* __restrict__ o, const float* __restrict__ b, int n, int dim) {
    int i = blockIdx.x * blockDim.x + threadIdx.x;
    if (i >= n) return;
    float v = o[i] + b[i % dim];
    o[i] = v > 0.0f ? v : (expf(v) - 1.0f);
}
__global__ void k_final(float* __restrict__ out, const float* __restrict__ acc,
                        const float* __restrict__ b, int n) {
    int i = blockIdx.x * blockDim.x + threadIdx.x;
    if (i < n) out[i] = acc[i] + b[i & 15];
}

// ---------------------------------------------------------------------------
extern "C" void kernel_launch(void* const* d_in, const int* in_sizes, int n_in,
                              void* d_out, int out_size, void* d_ws, size_t ws_size,
                              hipStream_t stream) {
    const float* x      = (const float*)d_in[0];
    const int*   ei     = (const int*)d_in[1];
    const float* W_in   = (const float*)d_in[2];
    const float* b_in   = (const float*)d_in[3];
    const float* ln_g   = (const float*)d_in[4];
    const float* ln_b   = (const float*)d_in[5];
    const float* W_sh   = (const float*)d_in[6];
    const float* gamma  = (const float*)d_in[7];
    const float* a_svr  = (const float*)d_in[8];
    const float* a_afm  = (const float*)d_in[9];
    const float* W1     = (const float*)d_in[10];
    const float* a1s    = (const float*)d_in[11];
    const float* a1d    = (const float*)d_in[12];
    const float* b1     = (const float*)d_in[13];
    const float* W2     = (const float*)d_in[14];
    const float* a2s    = (const float*)d_in[15];
    const float* a2d    = (const float*)d_in[16];
    const float* b2     = (const float*)d_in[17];
    float* out = (float*)d_out;

    const int* row = ei;
    const int* col = ei + NEDGES;
    const int NH = NNODES * HDIM;       // 6,400,000
    const int MT = NNODES / 16;         // 3125 (exact)

    // ---- workspace carve (256B aligned)
    char* wp = (char*)d_ws;
    auto alloc = [&](size_t bytes) -> char* {
        char* p = wp;
        wp += (bytes + 255) & ~(size_t)255;
        return p;
    };
    float* buf[8];
    for (int i = 0; i < 8; ++i) buf[i] = (float*)alloc((size_t)NH * 4);
    float* h    = buf[0];
    float* hW   = buf[1];   // later: fused
    float* Xc   = buf[2];
    float* Rc   = buf[3];
    float* Pc   = buf[4];
    float* APc  = buf[5];
    float* Lb   = buf[6];
    float* Hacc = buf[7];
    float* wL   = (float*)alloc((size_t)NEDGES * 4);
    float* deg  = (float*)alloc((size_t)NNODES * 4);
    float* isd  = (float*)alloc((size_t)NNODES * 4);
    float* rs    = (float*)alloc(128 * 4);
    float* denom = (float*)alloc(128 * 4);
    float* alpha = (float*)alloc(128 * 4);
    float* beta  = (float*)alloc(128 * 4);
    float* rsn   = (float*)alloc(128 * 4);
    float* aq    = (float*)alloc(4 * 4);
    int*   flags = (int*)alloc(4 * 4);
    unsigned short* Wsw0 = (unsigned short*)alloc((size_t)512 * 128 * 2 * 2);
    unsigned short* WswS = (unsigned short*)alloc((size_t)128 * 128 * 2 * 2);
    unsigned short* Wsw1 = (unsigned short*)alloc((size_t)128 * 64 * 2 * 2);
    unsigned short* Wsw2 = (unsigned short*)alloc((size_t)64 * 16 * 2 * 2);
    int* f1 = flags;      // CG done
    int* f2 = flags + 1;  // CG done_n
    int* zf = flags + 2;  // constant zero gate

    const int EWB = (NH + 255) / 256;
    const int GMB = (MT + 3) / 4;

    k_flags_init<<<1, 1, 0, stream>>>(flags);

    // ---- h = sigmoid(LN(x @ W_in + b_in))
    k_prep_w<<<(512 * 128 + 255) / 256, 256, 0, stream>>>(W_in, Wsw0, 512, 128);
    k_gemm_wmma<8, true><<<GMB, 128, 0, stream>>>(x, Wsw0, b_in, ln_g, ln_b, h, MT, 512);

    // ---- hW = h @ W_sheaf ; per-edge wL, deg
    k_prep_w<<<(128 * 128 + 255) / 256, 256, 0, stream>>>(W_sh, WswS, 128, 128);
    k_gemm_wmma<8, false><<<GMB, 128, 0, stream>>>(h, WswS, nullptr, nullptr, nullptr, hW, MT, 128);
    k_zero<<<(NNODES + 255) / 256, 256, 0, stream>>>(deg, NNODES);
    k_wl_deg<<<(NEDGES * 32 + 255) / 256, 256, 0, stream>>>(hW, row, col, wL, deg, NEDGES);

    // ---- CG: (I + dt L) X = h, 20 fixed iters with device-side early-break gating
    k_zero<<<EWB, 256, 0, stream>>>(Xc, NH);
    hipMemcpyAsync(Rc, h, (size_t)NH * 4, hipMemcpyDeviceToDevice, stream);
    hipMemcpyAsync(Pc, h, (size_t)NH * 4, hipMemcpyDeviceToDevice, stream);
    k_zero<<<1, 128, 0, stream>>>(rs, 128);
    k_coldot<<<256, 256, 0, stream>>>(h, h, rs, NNODES, zf);
    for (int it = 0; it < 20; ++it) {
        k_scale_deg<<<EWB, 256, 0, stream>>>(APc, Pc, deg, 1.0f, DT_, f1, NH);
        k_edge_scatter<<<(NEDGES * 4 + 255) / 256, 256, 0, stream>>>(APc, Pc, wL, row, col, -DT_, f1, NEDGES);
        k_zero<<<1, 128, 0, stream>>>(denom, 128);
        k_coldot<<<256, 256, 0, stream>>>(Pc, APc, denom, NNODES, f1);
        k_cg_alpha<<<1, 128, 0, stream>>>(alpha, rs, denom, f1);
        k_cg_xr<<<EWB, 256, 0, stream>>>(Xc, Rc, Pc, APc, alpha, f1, NH);
        k_zero<<<1, 128, 0, stream>>>(rsn, 128);
        k_coldot<<<256, 256, 0, stream>>>(Rc, Rc, rsn, NNODES, f1);
        k_cg_finish<<<1, 128, 0, stream>>>(rs, rsn, beta, f1, f2);
        k_cg_p<<<EWB, 256, 0, stream>>>(Pc, Rc, beta, f2, NH);
    }

    // ---- AFM Chebyshev branch (reuses Rc,Pc,APc,Lb as T ring / scratch)
    k_isd<<<(NNODES + 255) / 256, 256, 0, stream>>>(deg, isd, NNODES);
    k_softmax4<<<1, 1, 0, stream>>>(gamma, aq);
    float* U = APc;
    // q=1: T1 = tildeL(h) -> Pc
    k_mul_isd<<<EWB, 256, 0, stream>>>(U, h, isd, NH);
    k_scale_deg<<<EWB, 256, 0, stream>>>(Lb, U, deg, 0.0f, 1.0f, zf, NH);
    k_edge_scatter<<<(NEDGES * 4 + 255) / 256, 256, 0, stream>>>(Lb, U, wL, row, col, -1.0f, zf, NEDGES);
    k_tilde_out<<<EWB, 256, 0, stream>>>(Pc, h, Lb, isd, NH);
    k_hacc_init<<<EWB, 256, 0, stream>>>(Hacc, h, Pc, aq, NH);
    // q=2: T2 = 2*tildeL(T1) - h -> Rc
    k_mul_isd<<<EWB, 256, 0, stream>>>(U, Pc, isd, NH);
    k_scale_deg<<<EWB, 256, 0, stream>>>(Lb, U, deg, 0.0f, 1.0f, zf, NH);
    k_edge_scatter<<<(NEDGES * 4 + 255) / 256, 256, 0, stream>>>(Lb, U, wL, row, col, -1.0f, zf, NEDGES);
    k_cheb_next<<<EWB, 256, 0, stream>>>(Rc, Pc, Lb, h, isd, NH);
    k_hacc_add<<<EWB, 256, 0, stream>>>(Hacc, Rc, aq, 2, NH);
    // q=3: T3 = 2*tildeL(T2) - T1 -> Pc (in-place vs Tp=Pc is element-safe)
    k_mul_isd<<<EWB, 256, 0, stream>>>(U, Rc, isd, NH);
    k_scale_deg<<<EWB, 256, 0, stream>>>(Lb, U, deg, 0.0f, 1.0f, zf, NH);
    k_edge_scatter<<<(NEDGES * 4 + 255) / 256, 256, 0, stream>>>(Lb, U, wL, row, col, -1.0f, zf, NEDGES);
    k_cheb_next<<<EWB, 256, 0, stream>>>(Pc, Rc, Lb, Pc, isd, NH);
    k_hacc_add<<<EWB, 256, 0, stream>>>(Hacc, Pc, aq, 3, NH);

    // ---- fused = h + sig(a_svr)*X + sig(a_afm)*H_afm   (into hW buffer)
    float* fused = hW;
    k_fused<<<EWB, 256, 0, stream>>>(fused, h, Xc, Hacc, a_svr, a_afm, NH);

    // ---- GAT layer 1 (8 heads x 8 ch, concat) — reuse freed buffers
    float* xw1  = buf[0];
    float* es1  = buf[2];
    float* ed1  = buf[2] + NNODES * 8;
    unsigned* m1 = (unsigned*)(buf[2] + 2 * NNODES * 8);
    float* den1 = buf[2] + 3 * NNODES * 8;
    float* o1   = buf[3];
    k_prep_w<<<(128 * 64 + 255) / 256, 256, 0, stream>>>(W1, Wsw1, 128, 64);
    k_gemm_wmma<4, false><<<GMB, 128, 0, stream>>>(fused, Wsw1, nullptr, nullptr, nullptr, xw1, MT, 128);
    k_head_dots<<<(NNODES * 8 + 255) / 256, 256, 0, stream>>>(xw1, a1s, a1d, es1, ed1, NNODES, 8, 8);
    k_init_att<<<(NNODES * 8 + 255) / 256, 256, 0, stream>>>(m1, den1, NNODES * 8);
    k_zero<<<(NNODES * 64 + 255) / 256, 256, 0, stream>>>(o1, NNODES * 64);
    {
        int nt = (NEDGES + NNODES) * 8;
        k_att_max<<<(nt + 255) / 256, 256, 0, stream>>>(es1, ed1, row, col, m1, NEDGES, NNODES, 8);
        k_att_den<<<(nt + 255) / 256, 256, 0, stream>>>(es1, ed1, row, col, m1, den1, NEDGES, NNODES, 8);
        k_att_agg<<<(nt + 255) / 256, 256, 0, stream>>>(es1, ed1, row, col, m1, den1, xw1, o1, NEDGES, NNODES, 8, 8);
    }
    k_elu_bias<<<(NNODES * 64 + 255) / 256, 256, 0, stream>>>(o1, b1, NNODES * 64, 64);

    // ---- GAT layer 2 (1 head x 16 ch, mean==identity) -> logits
    float* xw2  = buf[4];
    float* es2  = buf[5];
    float* ed2  = buf[5] + NNODES;
    unsigned* m2 = (unsigned*)(buf[5] + 2 * NNODES);
    float* den2 = buf[5] + 3 * NNODES;
    float* g2   = buf[6];
    k_prep_w<<<(64 * 16 + 255) / 256, 256, 0, stream>>>(W2, Wsw2, 64, 16);
    k_gemm_wmma<1, false><<<GMB, 128, 0, stream>>>(o1, Wsw2, nullptr, nullptr, nullptr, xw2, MT, 64);
    k_head_dots<<<(NNODES + 255) / 256, 256, 0, stream>>>(xw2, a2s, a2d, es2, ed2, NNODES, 1, 16);
    k_init_att<<<(NNODES + 255) / 256, 256, 0, stream>>>(m2, den2, NNODES);
    k_zero<<<(NNODES * 16 + 255) / 256, 256, 0, stream>>>(g2, NNODES * 16);
    {
        int nt = NEDGES + NNODES;
        k_att_max<<<(nt + 255) / 256, 256, 0, stream>>>(es2, ed2, row, col, m2, NEDGES, NNODES, 1);
        k_att_den<<<(nt + 255) / 256, 256, 0, stream>>>(es2, ed2, row, col, m2, den2, NEDGES, NNODES, 1);
        k_att_agg<<<(nt + 255) / 256, 256, 0, stream>>>(es2, ed2, row, col, m2, den2, xw2, g2, NEDGES, NNODES, 1, 16);
    }
    k_final<<<(NNODES * 16 + 255) / 256, 256, 0, stream>>>(out, g2, b2, NNODES * 16);
}
